// SNNModel_34737695490053
// MI455X (gfx1250) — matrix operand order repaired and checked by reading, not compiled, not run on hardware
//
#include <hip/hip_runtime.h>

// ---- problem constants (from reference) ----
#define BATCH   32
#define T_LEN   1024
#define N_INP   128
#define N_HID   1024
#define KSLOT   8
#define DMAX    50
#define NDELAY  51          // d = 0..50
#define ALPHA   0.1f        // DT / TAU_M
#define VTH     1.0f
#define REFRAC  2.0f

typedef _Float16 v16h __attribute__((ext_vector_type(16)));
typedef _Float16 h8   __attribute__((ext_vector_type(8)));
typedef float    v8f  __attribute__((ext_vector_type(8)));

#define HTILES  (N_HID / 16)          // 64
#define KCH     (N_INP / 32)          // 4
#define XH_BYTES  ((size_t)BATCH * T_LEN * N_INP * 2)                  // 8 MB
#define WP_HALFS  ((size_t)NDELAY * HTILES * KCH * 32 * 16)            // 6.68M halfs
#define WP_BYTES  (WP_HALFS * 2)                                       // 13.4 MB

// ------------------------------------------------------------------
// Convert 0/1 spike input to fp16 (exact).
// ------------------------------------------------------------------
__global__ void snn_convert_x(const float* __restrict__ xin,
                              _Float16* __restrict__ xh, int n) {
    int i = blockIdx.x * blockDim.x + threadIdx.x;
    if (i < n) xh[i] = (_Float16)xin[i];
}

// ------------------------------------------------------------------
// Pack delay-masked weights W_d into B-operand layout:
//   wp[((d*64 + ht)*4 + kc)*32 + lane][16 halfs]
//   lane: n = lane&15 (column h = ht*16+n), g = lane>>4
//   half j -> K = kc*32 + g*16 + j   (contiguous 16 K per lane)
// ------------------------------------------------------------------
__global__ void snn_pack_w(const float* __restrict__ W,
                           const float* __restrict__ draw,
                           _Float16* __restrict__ wp) {
    int idx = blockIdx.x * blockDim.x + threadIdx.x;   // (d, ht, kc, lane)
    if (idx >= NDELAY * HTILES * KCH * 32) return;
    int lane = idx & 31;
    int kc   = (idx >> 5) & 3;
    int ht   = (idx >> 7) & 63;
    int d    = idx >> 13;
    int n = lane & 15, g = lane >> 4;
    int h = ht * 16 + n;
    h8 lo = {}, hi = {};
    #pragma unroll
    for (int j = 0; j < 16; ++j) {
        int i = kc * 32 + g * 16 + j;
        float dr  = draw[h * N_INP + i];
        float sig = 1.0f / (1.0f + expf(-dr));
        int   di  = (int)rintf(sig * (float)DMAX);   // round-half-even like jnp.round
        _Float16 v = (di == d) ? (_Float16)W[h * N_INP + i] : (_Float16)0.0f;
        if (j < 8) lo[j] = v; else hi[j - 8] = v;
    }
    _Float16* dst = wp + (size_t)idx * 16;
    *(h8*)(dst)     = lo;
    *(h8*)(dst + 8) = hi;
}

// ------------------------------------------------------------------
// Seed output: logits = b_ro, totals = 0 (rewritten every call).
// ------------------------------------------------------------------
__global__ void snn_init_out(float* __restrict__ out, const float* __restrict__ b_ro) {
    int i = blockIdx.x * blockDim.x + threadIdx.x;
    if (i < BATCH * KSLOT)              out[i] = b_ro[0];
    else if (i < BATCH * KSLOT + T_LEN) out[i] = 0.0f;
}

// ------------------------------------------------------------------
// Phase A: I[b,t,h] = sum_d X[b, t-1-d, :] . W_d[h, :]
// Block = (t-tile of 16, batch b); 4 waves; wave handles 16 h-tiles,
// 4 accumulators in flight per A load. WMMA f32 = f16 x f16.
// ------------------------------------------------------------------
__device__ __forceinline__ v16h snn_load_a(const _Float16* sX, int lane, int d, int kc) {
    int m = lane & 15, g = lane >> 4;
    int r  = m + 50 - d;                     // row in staged window
    int k1 = kc * 32 + g * 16;               // A layout: contiguous 16 K at K=g*16
    h8 lo = *(const h8*)(sX + r * N_INP + k1);
    h8 hi = *(const h8*)(sX + r * N_INP + k1 + 8);
    return __builtin_shufflevector(lo, hi, 0,1,2,3,4,5,6,7,8,9,10,11,12,13,14,15);
}

__device__ __forceinline__ v16h snn_load_b(const _Float16* __restrict__ wp,
                                           int lane, int d, int ht, int kc) {
    const _Float16* p = wp + ((((size_t)d * HTILES + ht) * KCH + kc) * 32 + lane) * 16;
    h8 lo = *(const h8*)p;
    h8 hi = *(const h8*)(p + 8);
    return __builtin_shufflevector(lo, hi, 0,1,2,3,4,5,6,7,8,9,10,11,12,13,14,15);
}

__global__ __launch_bounds__(128) void snn_gemm(const _Float16* __restrict__ xh,
                                                const _Float16* __restrict__ wp,
                                                float* __restrict__ Ibuf) {
    __shared__ _Float16 sX[66 * N_INP];      // rows t0-51 .. t0+14
    const int b  = blockIdx.y;
    const int t0 = blockIdx.x * 16;
    const int tid = threadIdx.x;

    // Stage input window into LDS (16B vectors; a vector never crosses rows).
    for (int i = tid; i < 66 * N_INP / 8; i += 128) {
        int r  = i >> 4;
        int c8 = (i & 15) * 8;
        int t  = t0 - 51 + r;
        h8 v = {};
        if (t >= 0) v = *(const h8*)(xh + ((size_t)b * T_LEN + t) * N_INP + c8);
        *(h8*)(sX + r * N_INP + c8) = v;
    }
    __syncthreads();

    const int lane = tid & 31;
    const int warp = tid >> 5;

    for (int hg = 0; hg < 4; ++hg) {
        const int ht = warp * 16 + hg * 4;   // this wave's 4 h-tiles
        v8f c0 = {}, c1 = {}, c2 = {}, c3 = {};
        #pragma unroll 1
        for (int d = 0; d < NDELAY; ++d) {
            // Prefetch next delay's B block (16 KB per h-tile group) into near cache.
            if (d + 1 < NDELAY) {
                const _Float16* pf =
                    wp + ((((size_t)(d + 1) * HTILES + ht) * KCH) * 32 + lane) * 16;
                __builtin_prefetch(pf, 0, 0);
            }
            #pragma unroll
            for (int kc = 0; kc < KCH; ++kc) {
                // Issue all loads up front so WMMAs overlap in-flight B loads
                // (loads complete in order -> partial s_wait_loadcnt).
                v16h a  = snn_load_a(sX, lane, d, kc);
                v16h b0 = snn_load_b(wp, lane, d, ht + 0, kc);
                v16h b1 = snn_load_b(wp, lane, d, ht + 1, kc);
                v16h b2 = snn_load_b(wp, lane, d, ht + 2, kc);
                v16h b3 = snn_load_b(wp, lane, d, ht + 3, kc);
                c0 = __builtin_amdgcn_wmma_f32_16x16x32_f16(false, a, false, b0,
                                                            (short)0, c0, false, false);
                c1 = __builtin_amdgcn_wmma_f32_16x16x32_f16(false, a, false, b1,
                                                            (short)0, c1, false, false);
                c2 = __builtin_amdgcn_wmma_f32_16x16x32_f16(false, a, false, b2,
                                                            (short)0, c2, false, false);
                c3 = __builtin_amdgcn_wmma_f32_16x16x32_f16(false, a, false, b3,
                                                            (short)0, c3, false, false);
            }
        }
        // C layout: lane n = col h, VGPR v -> row m = g*8 + v
        const int n = lane & 15, g = lane >> 4;
        #pragma unroll
        for (int v = 0; v < 8; ++v) {
            int t = t0 + g * 8 + v;
            float* base = Ibuf + ((size_t)b * T_LEN + t) * N_HID + n;
            base[(ht + 0) * 16] = c0[v];
            base[(ht + 1) * 16] = c1[v];
            base[(ht + 2) * 16] = c2[v];
            base[(ht + 3) * 16] = c3[v];
        }
    }
}

// ------------------------------------------------------------------
// Phase B: LIF scan. One thread per (b,h); ballot+popc for per-t
// population counts into LDS, shfl reduction for readout.
// ------------------------------------------------------------------
__global__ __launch_bounds__(256) void snn_lif(const float* __restrict__ Ibuf,
                                               const float* __restrict__ w_ro,
                                               const int* __restrict__ sstart,
                                               const int* __restrict__ send,
                                               float* __restrict__ out) {
    __shared__ float spk[T_LEN];
    const int gid = blockIdx.x * 256 + threadIdx.x;
    const int b = gid >> 10;
    const int h = gid & (N_HID - 1);

    for (int i = threadIdx.x; i < T_LEN; i += 256) spk[i] = 0.0f;
    __syncthreads();

    int st[KSLOT], en[KSLOT];
    #pragma unroll
    for (int k = 0; k < KSLOT; ++k) {
        st[k] = sstart[k];
        int e = send[k];
        en[k] = e < T_LEN ? e : T_LEN;
    }

    float v = 0.0f, ref = 0.0f;
    float cnt[KSLOT];
    #pragma unroll
    for (int k = 0; k < KSLOT; ++k) cnt[k] = 0.0f;

    const float* Ib = Ibuf + ((size_t)b * T_LEN) * N_HID + h;
    for (int t = 0; t < T_LEN; ++t) {
        float Icur = Ib[(size_t)t * N_HID];
        bool active = ref <= 0.0f;
        float vu = fmaf(ALPHA, Icur - v, v);          // v + alpha*(-v + I)
        float vn = active ? vu : v;
        bool s = active && (vn - VTH >= 0.0f);
        if (s) { vn = 0.0f; ref = REFRAC; }
        else   { ref = fmaxf(ref - 1.0f, 0.0f); }
        v = vn;

        unsigned long long bal = __ballot(s ? 1 : 0);
        if ((threadIdx.x & 31) == 0) {
            float c = (float)__popcll(bal);
            if (c > 0.0f) atomicAdd(&spk[t], c);      // ds_add_f32
        }
        if (s) {
            #pragma unroll
            for (int k = 0; k < KSLOT; ++k)
                if (t >= st[k] && t < en[k]) cnt[k] += 1.0f;
        }
    }
    __syncthreads();

    float* total = out + BATCH * KSLOT;               // out layout: [logits(256) | totals(1024)]
    for (int i = threadIdx.x; i < T_LEN; i += 256)
        atomicAdd(&total[i], spk[i]);

    const float wro = w_ro[h];
    #pragma unroll
    for (int k = 0; k < KSLOT; ++k) {
        float val = cnt[k] * wro;
        for (int off = 16; off > 0; off >>= 1)
            val += __shfl_down(val, off, 32);
        if ((threadIdx.x & 31) == 0)
            atomicAdd(&out[b * KSLOT + k], val);
    }
}

// ------------------------------------------------------------------
extern "C" void kernel_launch(void* const* d_in, const int* in_sizes, int n_in,
                              void* d_out, int out_size, void* d_ws, size_t ws_size,
                              hipStream_t stream) {
    const float* spike_input = (const float*)d_in[0];
    const float* W_ih        = (const float*)d_in[1];
    const float* delay_raw   = (const float*)d_in[2];
    const float* w_ro        = (const float*)d_in[3];
    const float* b_ro        = (const float*)d_in[4];
    const int*   ss          = (const int*)d_in[5];
    const int*   se          = (const int*)d_in[6];
    (void)in_sizes; (void)n_in; (void)out_size; (void)ws_size;

    char* ws = (char*)d_ws;
    _Float16* xh   = (_Float16*)ws;                          // 8 MB
    _Float16* wp   = (_Float16*)(ws + XH_BYTES);             // 13.4 MB
    float*    Ibuf = (float*)(ws + XH_BYTES + WP_BYTES);     // 128 MB

    const int nx = BATCH * T_LEN * N_INP;
    snn_convert_x<<<(nx + 255) / 256, 256, 0, stream>>>(spike_input, xh, nx);

    const int np = NDELAY * HTILES * KCH * 32;
    snn_pack_w<<<(np + 255) / 256, 256, 0, stream>>>(W_ih, delay_raw, wp);

    snn_init_out<<<(BATCH * KSLOT + T_LEN + 255) / 256, 256, 0, stream>>>((float*)d_out, b_ro);

    dim3 g(T_LEN / 16, BATCH);                               // (64, 32) blocks
    snn_gemm<<<g, 128, 0, stream>>>(xh, wp, Ibuf);

    snn_lif<<<(BATCH * N_HID) / 256, 256, 0, stream>>>(Ibuf, w_ro, ss, se, (float*)d_out);
}
